// NSTB_27470610825464
// MI455X (gfx1250) — matrix operand order repaired and checked by reading, not compiled, not run on hardware
//
#include <hip/hip_runtime.h>
#include <math.h>

// ---------------- types / helpers ----------------
typedef __bf16 bf16;
typedef __attribute__((ext_vector_type(16))) bf16  v16bf;
typedef __attribute__((ext_vector_type(8)))  float v8f;

__device__ inline bf16 f2bf(float f) {
    unsigned u = __builtin_bit_cast(unsigned, f);
    u += 0x7FFFu + ((u >> 16) & 1u);            // round-to-nearest-even
    unsigned short s = (unsigned short)(u >> 16);
    return __builtin_bit_cast(bf16, s);
}

// A fragment (16x32 bf16, M x K): lane&15 = row, halves 0-7 = k+{0..7}, 8-15 = k+16+{0..7},
// upper half-wave shifts k by +8.  (ISA 7.12.2, 16-bit A 16x32)
__device__ inline v16bf ldfragA(const bf16* base, int ld, int row, int kb) {
    int lane = threadIdx.x & 31;
    const bf16* p = base + (row + (lane & 15)) * ld + kb + ((lane >> 4) << 3);
    v16bf r;
#pragma unroll
    for (int i = 0; i < 8; ++i) { r[i] = p[i]; r[i + 8] = p[i + 16]; }
    return r;
}

// B fragment (32x16 bf16, K x N) from a row-major [N][K] view (i.e. B^T):
// lane&15 = column n, halves 0-7 = k+{0..7}, 8-15 = k+16+{0..7}, upper half-wave k+8.
__device__ inline v16bf ldfragB(const bf16* base, int ld, int col, int kb) {
    int lane = threadIdx.x & 31;
    const bf16* p = base + (col + (lane & 15)) * ld + kb + ((lane >> 4) << 3);
    v16bf r;
#pragma unroll
    for (int i = 0; i < 8; ++i) { r[i] = p[i]; r[i + 8] = p[i + 16]; }
    return r;
}

__device__ inline v8f wmma_bf16(v16bf a, v16bf b, v8f c) {
    return __builtin_amdgcn_wmma_f32_16x16x32_bf16(false, a, false, b, (short)0, c, false, false);
}

__device__ inline float wred_sum(float v) {
#pragma unroll
    for (int m = 16; m >= 1; m >>= 1) v += __shfl_xor(v, m, 32);
    return v;
}
__device__ inline float wred_max(float v) {
#pragma unroll
    for (int m = 16; m >= 1; m >>= 1) v = fmaxf(v, __shfl_xor(v, m, 32));
    return v;
}

__device__ inline int grp(int p) { return p < 248 ? 0 : (p < 252 ? 1 : 2); }

// ---------------- weight f32 -> bf16 ----------------
__global__ void k_cvt_bf16(const float* __restrict__ s, bf16* __restrict__ d, int n) {
    int i = blockIdx.x * blockDim.x + threadIdx.x;
    if (i < n) d[i] = f2bf(s[i]);
}

// ---------------- grouped 8x8/s8 conv: uni[b][oh][ow][c], c<64 ----------------
__global__ void k_uni(const float* __restrict__ x, const float* __restrict__ w,
                      const float* __restrict__ bias, float* __restrict__ uni) {
    int gid = blockIdx.x * blockDim.x + threadIdx.x;   // 4*32*32*64
    int c = gid & 63, ow = (gid >> 6) & 31, oh = (gid >> 11) & 31, b = gid >> 16;
    float acc = bias[c];
#pragma unroll
    for (int ic2 = 0; ic2 < 2; ++ic2) {
        int ic = 2 * c + ic2;
        const float* wp = w + (c * 2 + ic2) * 64;
        for (int kh = 0; kh < 8; ++kh)
#pragma unroll
            for (int kw = 0; kw < 8; ++kw) {
                int pos = (oh * 8 + kh) * 256 + (ow * 8 + kw);
                acc += x[(b * 65536 + pos) * 128 + ic] * wp[kh * 8 + kw];
            }
    }
    uni[gid] = acc;
}

// ---------------- ngram sliding attention; one 64-thread block per (dir,b,i,j) window ----------------
__global__ void k_ngram(const float* __restrict__ uni,
                        const float* __restrict__ qkvw, const float* __restrict__ qkvb,
                        const float* __restrict__ projw, const float* __restrict__ projb,
                        const float* __restrict__ btab, float* __restrict__ ctx) {
    __shared__ float xw[16][64], Ks[16][64], Vs[16][64], Os[16][64];
    int blk = blockIdx.x;                 // dir*4096 + b*1024 + i*32 + j
    int j = blk & 31, i = (blk >> 5) & 31, b = (blk >> 10) & 3, dir = blk >> 12;
    int tid = threadIdx.x;
    for (int idx = tid; idx < 1024; idx += 64) {
        int t = idx >> 6, c = idx & 63;
        int pi = i + (t >> 2), pj = j + (t & 3);
        int mi = dir ? (pi < 3 ? pi + 1 : pi - 3) : (pi < 32 ? pi : pi - 4);
        int mj = dir ? (pj < 3 ? pj + 1 : pj - 3) : (pj < 32 ? pj : pj - 4);
        xw[t][c] = uni[((b * 32 + mi) * 32 + mj) * 64 + c];
    }
    __syncthreads();
    int t = tid >> 2, h = tid & 3;
    float q[16];
#pragma unroll 4
    for (int e = 0; e < 16; ++e) {
        int rq = h * 16 + e;
        float aq = qkvb[rq], ak = qkvb[64 + rq], av = qkvb[128 + rq];
        const float* wq = qkvw + rq * 64;
        const float* wk = qkvw + (64 + rq) * 64;
        const float* wv = qkvw + (128 + rq) * 64;
        for (int c = 0; c < 64; ++c) {
            float xv = xw[t][c];
            aq += xv * wq[c]; ak += xv * wk[c]; av += xv * wv[c];
        }
        q[e] = aq * 0.25f;                 // hd=16 -> 16^-0.5
        Ks[t][h * 16 + e] = ak;
        Vs[t][h * 16 + e] = av;
    }
    __syncthreads();
    float s[16], mx = -1e30f;
    int i1 = t >> 2, j1 = t & 3;
#pragma unroll
    for (int m = 0; m < 16; ++m) {
        float a = 0.f;
#pragma unroll
        for (int e = 0; e < 16; ++e) a += q[e] * Ks[m][h * 16 + e];
        int i2 = m >> 2, j2 = m & 3;
        a += btab[((i1 - i2 + 3) * 7 + (j1 - j2 + 3)) * 4 + h];
        s[m] = a; mx = fmaxf(mx, a);
    }
    float den = 0.f;
#pragma unroll
    for (int m = 0; m < 16; ++m) { s[m] = __expf(s[m] - mx); den += s[m]; }
    float inv = 1.f / den;
#pragma unroll
    for (int e = 0; e < 16; ++e) {
        float o = 0.f;
#pragma unroll
        for (int m = 0; m < 16; ++m) o += s[m] * Vs[m][h * 16 + e];
        Os[t][h * 16 + e] = o * inv;
    }
    __syncthreads();
    int d = tid;                           // proj + avg-pool(NG) == mean over tokens
    float acc = 0.f;
    const float* wp = projw + d * 64;
    for (int tt = 0; tt < 16; ++tt) {
        float y = 0.f;
        for (int c = 0; c < 64; ++c) y += Os[tt][c] * wp[c];
        acc += y;
    }
    ctx[((b * 32 + i) * 32 + j) * 128 + dir * 64 + d] = acc * (1.f / 16.f) + projb[d];
}

// ---------------- 1x1 merge conv, channel-last ctx ----------------
__global__ void k_merge(const float* __restrict__ ctx, const float* __restrict__ mw,
                        const float* __restrict__ mb, float* __restrict__ ctx2) {
    int gid = blockIdx.x * blockDim.x + threadIdx.x;   // 4*32*32*128
    int o = gid & 127;
    const float* cp = ctx + (gid >> 7) * 128;
    const float* wp = mw + o * 128;
    float acc = mb[o];
    for (int c = 0; c < 128; ++c) acc += cp[c] * wp[c];
    ctx2[gid] = acc;
}

// ---------------- masked window MHSA, WMMA bf16; fused ctx-inject + cyclic shift ----------------
// reads x (tokens) + ctx2, writes xr = attention output in shifted-window layout
__global__ __launch_bounds__(256) void k_wattn(const float* __restrict__ x,
        const float* __restrict__ ctx2, float* __restrict__ xr,
        const bf16* __restrict__ qkvw, const float* __restrict__ qkvb,
        const bf16* __restrict__ projw, const float* __restrict__ projb,
        const float* __restrict__ btab) {
    __shared__ bf16  XO[64][128];      // X during QKV, attention output O afterwards
    __shared__ bf16  Qs[64][128];
    __shared__ bf16  Ksm[64][128];
    __shared__ bf16  Vt[128][64];      // V transposed: [channel][token]
    __shared__ float Ss[64][64];
    __shared__ bf16  Ps[64][64];
    int blk = blockIdx.x;
    int wj = blk & 31, wi = (blk >> 5) & 31, b = blk >> 10;
    int tid = threadIdx.x, wave = tid >> 5, lane = tid & 31;

    // stage 0: fused  X = roll(x + ctx_broadcast)  -> LDS bf16
    for (int idx = tid; idx < 2048; idx += 256) {       // float4 granules
        int t = idx >> 5, d4 = idx & 31;
        int h = wi * 8 + (t >> 3), w = wj * 8 + (t & 7);
        int hh = (h + 4) & 255, ww = (w + 4) & 255;
        float4 xv = reinterpret_cast<const float4*>(x)[(b * 65536 + hh * 256 + ww) * 32 + d4];
        float4 cv = reinterpret_cast<const float4*>(ctx2)[((b * 32 + (hh >> 3)) * 32 + (ww >> 3)) * 32 + d4];
        XO[t][d4 * 4 + 0] = f2bf(xv.x + cv.x);
        XO[t][d4 * 4 + 1] = f2bf(xv.y + cv.y);
        XO[t][d4 * 4 + 2] = f2bf(xv.z + cv.z);
        XO[t][d4 * 4 + 3] = f2bf(xv.w + cv.w);
    }
    __syncthreads();

    // ---- Q tiles (uniform epilogue, scale folded in) ----
    for (int tk = wave; tk < 32; tk += 8) {
        int mt = tk & 3, nt = tk >> 2;
        v8f acc = {0.f,0.f,0.f,0.f,0.f,0.f,0.f,0.f};
#pragma unroll
        for (int ks = 0; ks < 4; ++ks) {
            v16bf a  = ldfragA(&XO[0][0], 128, mt * 16, ks * 32);
            v16bf bm = ldfragB(qkvw, 128, nt * 16, ks * 32);
            acc = wmma_bf16(a, bm, acc);
        }
        int n = nt * 16 + (lane & 15);
        float bias = qkvb[n];
#pragma unroll
        for (int r = 0; r < 8; ++r) {
            int m = mt * 16 + r + ((lane >> 4) << 3);
            Qs[m][n] = f2bf((acc[r] + bias) * 0.1767766953f);   // hd=32 -> 32^-0.5
        }
    }
    // ---- K tiles ----
    for (int tk = wave; tk < 32; tk += 8) {
        int mt = tk & 3, nt = tk >> 2;
        v8f acc = {0.f,0.f,0.f,0.f,0.f,0.f,0.f,0.f};
#pragma unroll
        for (int ks = 0; ks < 4; ++ks) {
            v16bf a  = ldfragA(&XO[0][0], 128, mt * 16, ks * 32);
            v16bf bm = ldfragB(qkvw, 128, 128 + nt * 16, ks * 32);
            acc = wmma_bf16(a, bm, acc);
        }
        int n = nt * 16 + (lane & 15);
        float bias = qkvb[128 + n];
#pragma unroll
        for (int r = 0; r < 8; ++r) {
            int m = mt * 16 + r + ((lane >> 4) << 3);
            Ksm[m][n] = f2bf(acc[r] + bias);
        }
    }
    // ---- V tiles (stored transposed) ----
    for (int tk = wave; tk < 32; tk += 8) {
        int mt = tk & 3, nt = tk >> 2;
        v8f acc = {0.f,0.f,0.f,0.f,0.f,0.f,0.f,0.f};
#pragma unroll
        for (int ks = 0; ks < 4; ++ks) {
            v16bf a  = ldfragA(&XO[0][0], 128, mt * 16, ks * 32);
            v16bf bm = ldfragB(qkvw, 128, 256 + nt * 16, ks * 32);
            acc = wmma_bf16(a, bm, acc);
        }
        int n = nt * 16 + (lane & 15);
        float bias = qkvb[256 + n];
#pragma unroll
        for (int r = 0; r < 8; ++r) {
            int m = mt * 16 + r + ((lane >> 4) << 3);
            Vt[n][m] = f2bf(acc[r] + bias);
        }
    }
    __syncthreads();

    for (int head = 0; head < 4; ++head) {
        // S = Q_h @ K_h^T (K=32, one WMMA), + rel-pos bias + shift mask
        for (int tk = wave; tk < 16; tk += 8) {
            int mt = tk & 3, nt = tk >> 2;
            v16bf a  = ldfragA(&Qs[0][0], 128, mt * 16, head * 32);
            v16bf bm = ldfragB(&Ksm[0][0], 128, nt * 16, head * 32);
            v8f acc = {0.f,0.f,0.f,0.f,0.f,0.f,0.f,0.f};
            acc = wmma_bf16(a, bm, acc);
            int n = nt * 16 + (lane & 15);
            int i2 = n >> 3, j2 = n & 7;
            int g2 = grp(wi * 8 + i2) * 3 + grp(wj * 8 + j2);
#pragma unroll
            for (int r = 0; r < 8; ++r) {
                int m = mt * 16 + r + ((lane >> 4) << 3);
                int i1 = m >> 3, j1 = m & 7;
                float bias = btab[((i1 - i2 + 7) * 15 + (j1 - j2 + 7)) * 4 + head];
                int g1 = grp(wi * 8 + i1) * 3 + grp(wj * 8 + j1);
                Ss[m][n] = acc[r] + bias + ((g1 != g2) ? -100.f : 0.f);
            }
        }
        __syncthreads();
        // wave-parallel row softmax: 8 rows per wave, 2 cols per lane
        for (int rr = wave; rr < 64; rr += 8) {
            float a0 = Ss[rr][2 * lane], a1 = Ss[rr][2 * lane + 1];
            float mx = wred_max(fmaxf(a0, a1));
            float e0 = __expf(a0 - mx), e1 = __expf(a1 - mx);
            float inv = 1.f / wred_sum(e0 + e1);
            Ps[rr][2 * lane]     = f2bf(e0 * inv);
            Ps[rr][2 * lane + 1] = f2bf(e1 * inv);
        }
        __syncthreads();
        {   // O_h = P @ V_h : 8 tiles, one per wave; O aliases X (dead after QKV)
            int mt = wave & 3, nt = wave >> 2;
            v8f acc = {0.f,0.f,0.f,0.f,0.f,0.f,0.f,0.f};
#pragma unroll
            for (int ks = 0; ks < 2; ++ks) {
                v16bf a  = ldfragA(&Ps[0][0], 64, mt * 16, ks * 32);
                v16bf bm = ldfragB(&Vt[0][0], 64, head * 32 + nt * 16, ks * 32);
                acc = wmma_bf16(a, bm, acc);
            }
            int n = head * 32 + nt * 16 + (lane & 15);
#pragma unroll
            for (int r = 0; r < 8; ++r) {
                int m = mt * 16 + r + ((lane >> 4) << 3);
                XO[m][n] = f2bf(acc[r]);
            }
        }
        __syncthreads();
    }

    // proj: [64x128] @ [128x128]^T -> xr (shifted layout)
    for (int tk = wave; tk < 32; tk += 8) {
        int mt = tk & 3, nt = tk >> 2;
        v8f acc = {0.f,0.f,0.f,0.f,0.f,0.f,0.f,0.f};
#pragma unroll
        for (int ks = 0; ks < 4; ++ks) {
            v16bf a  = ldfragA(&XO[0][0], 128, mt * 16, ks * 32);
            v16bf bm = ldfragB(projw, 128, nt * 16, ks * 32);
            acc = wmma_bf16(a, bm, acc);
        }
        int n = nt * 16 + (lane & 15);
        float pb = projb[n];
#pragma unroll
        for (int r = 0; r < 8; ++r) {
            int m = mt * 16 + r + ((lane >> 4) << 3);
            int h = wi * 8 + (m >> 3), w = wj * 8 + (m & 7);
            xr[((b * 256 + h) * 256 + w) * 128 + n] = acc[r] + pb;
        }
    }
}

// ---------------- fused: unshift + LN1 + residual, FFN(WMMA), LN2 + residual ----------------
__global__ __launch_bounds__(256) void k_ffn(const float* __restrict__ x,
        const float* __restrict__ xr, float* __restrict__ out,
        const bf16* __restrict__ w1, const float* __restrict__ b1,
        const bf16* __restrict__ w2, const float* __restrict__ b2,
        const float* __restrict__ g1v, const float* __restrict__ b1v,
        const float* __restrict__ g2v, const float* __restrict__ b2v) {
    __shared__ float Xf[64][128];
    __shared__ bf16  Xh[64][128];
    __shared__ bf16  Hs[64][256];
    __shared__ float Fs[64][128];
    int tid = threadIdx.x, wave = tid >> 5, lane = tid & 31;
    int rowBase = blockIdx.x * 64;

    // stage 0: x1 = x + LN1(roll_back(attn)); one wave per row
    for (int rr = wave; rr < 64; rr += 8) {
        int row = rowBase + rr;
        int b = row >> 16, pos = row & 65535, h = pos >> 8, w = pos & 255;
        int sh = (h - 4) & 255, sw = (w - 4) & 255;
        float4 a = reinterpret_cast<const float4*>(xr)[((b * 256 + sh) * 256 + sw) * 32 + lane];
        float mu = wred_sum(a.x + a.y + a.z + a.w) * (1.f / 128.f);
        float dx = a.x - mu, dy = a.y - mu, dz = a.z - mu, dw = a.w - mu;
        float rs = rsqrtf(wred_sum(dx * dx + dy * dy + dz * dz + dw * dw) * (1.f / 128.f) + 1e-5f);
        float4 gv = reinterpret_cast<const float4*>(g1v)[lane];
        float4 bv = reinterpret_cast<const float4*>(b1v)[lane];
        float4 xv = reinterpret_cast<const float4*>(x)[row * 32 + lane];
        float4 x1;
        x1.x = xv.x + dx * rs * gv.x + bv.x;
        x1.y = xv.y + dy * rs * gv.y + bv.y;
        x1.z = xv.z + dz * rs * gv.z + bv.z;
        x1.w = xv.w + dw * rs * gv.w + bv.w;
        *reinterpret_cast<float4*>(&Xf[rr][lane * 4]) = x1;
        Xh[rr][lane * 4 + 0] = f2bf(x1.x);
        Xh[rr][lane * 4 + 1] = f2bf(x1.y);
        Xh[rr][lane * 4 + 2] = f2bf(x1.z);
        Xh[rr][lane * 4 + 3] = f2bf(x1.w);
    }
    __syncthreads();

    for (int tk = wave; tk < 64; tk += 8) {       // H = gelu(X @ W1^T + b1)  [64x256]
        int mt = tk & 3, nt = tk >> 2;
        v8f acc = {0.f,0.f,0.f,0.f,0.f,0.f,0.f,0.f};
#pragma unroll
        for (int ks = 0; ks < 4; ++ks) {
            v16bf a  = ldfragA(&Xh[0][0], 128, mt * 16, ks * 32);
            v16bf bm = ldfragB(w1, 128, nt * 16, ks * 32);
            acc = wmma_bf16(a, bm, acc);
        }
        int n = nt * 16 + (lane & 15);
        float bb = b1[n];
#pragma unroll
        for (int r = 0; r < 8; ++r) {
            int m = mt * 16 + r + ((lane >> 4) << 3);
            float v = acc[r] + bb;
            v = 0.5f * v * (1.f + erff(v * 0.70710678f));   // exact gelu
            Hs[m][n] = f2bf(v);
        }
    }
    __syncthreads();
    for (int tk = wave; tk < 32; tk += 8) {       // F = H @ W2^T + b2  [64x128]
        int mt = tk & 3, nt = tk >> 2;
        v8f acc = {0.f,0.f,0.f,0.f,0.f,0.f,0.f,0.f};
#pragma unroll
        for (int ks = 0; ks < 8; ++ks) {
            v16bf a  = ldfragA(&Hs[0][0], 256, mt * 16, ks * 32);
            v16bf bm = ldfragB(w2, 256, nt * 16, ks * 32);
            acc = wmma_bf16(a, bm, acc);
        }
        int n = nt * 16 + (lane & 15);
        float bb = b2[n];
#pragma unroll
        for (int r = 0; r < 8; ++r) {
            int m = mt * 16 + r + ((lane >> 4) << 3);
            Fs[m][n] = acc[r] + bb;
        }
    }
    __syncthreads();

    // out = x1 + LN2(F); one wave per row, coalesced float4 stores
    for (int rr = wave; rr < 64; rr += 8) {
        float4 f = *reinterpret_cast<const float4*>(&Fs[rr][lane * 4]);
        float mu = wred_sum(f.x + f.y + f.z + f.w) * (1.f / 128.f);
        float dx = f.x - mu, dy = f.y - mu, dz = f.z - mu, dw = f.w - mu;
        float rs = rsqrtf(wred_sum(dx * dx + dy * dy + dz * dz + dw * dw) * (1.f / 128.f) + 1e-5f);
        float4 gv = reinterpret_cast<const float4*>(g2v)[lane];
        float4 bv = reinterpret_cast<const float4*>(b2v)[lane];
        float4 xf = *reinterpret_cast<const float4*>(&Xf[rr][lane * 4]);
        float4 o;
        o.x = xf.x + dx * rs * gv.x + bv.x;
        o.y = xf.y + dy * rs * gv.y + bv.y;
        o.z = xf.z + dz * rs * gv.z + bv.z;
        o.w = xf.w + dw * rs * gv.w + bv.w;
        reinterpret_cast<float4*>(out)[(rowBase + rr) * 32 + lane] = o;
    }
}

// ---------------- launch ----------------
extern "C" void kernel_launch(void* const* d_in, const int* in_sizes, int n_in,
                              void* d_out, int out_size, void* d_ws, size_t ws_size,
                              hipStream_t stream) {
    const float* x        = (const float*)d_in[0];
    const float* uni_w    = (const float*)d_in[1];
    const float* uni_b    = (const float*)d_in[2];
    const float* ng_qkv_w = (const float*)d_in[3];
    const float* ng_qkv_b = (const float*)d_in[4];
    const float* ng_proj_w= (const float*)d_in[5];
    const float* ng_proj_b= (const float*)d_in[6];
    const float* ng_bias  = (const float*)d_in[7];
    const float* merge_w  = (const float*)d_in[8];
    const float* merge_b  = (const float*)d_in[9];
    const float* qkv_w    = (const float*)d_in[10];
    const float* qkv_b    = (const float*)d_in[11];
    const float* proj_w   = (const float*)d_in[12];
    const float* proj_b   = (const float*)d_in[13];
    const float* btab     = (const float*)d_in[14];
    const float* n1_g     = (const float*)d_in[15];
    const float* n1_b     = (const float*)d_in[16];
    const float* n2_g     = (const float*)d_in[17];
    const float* n2_b     = (const float*)d_in[18];
    const float* fc1_w    = (const float*)d_in[19];
    const float* fc1_b    = (const float*)d_in[20];
    const float* fc2_w    = (const float*)d_in[21];
    const float* fc2_b    = (const float*)d_in[22];
    (void)in_sizes; (void)n_in; (void)out_size; (void)ws_size;

    char* ws = (char*)d_ws;
    bf16*  qkvw_h = (bf16*)(ws);                    // 384*128 bf16
    bf16*  projw_h= (bf16*)(ws + 98304);            // 128*128 bf16
    bf16*  fc1w_h = (bf16*)(ws + 131072);           // 256*128 bf16
    bf16*  fc2w_h = (bf16*)(ws + 196608);           // 128*256 bf16
    float* uni    = (float*)(ws + 262144);          // 4*32*32*64 f32
    float* ctx    = (float*)(ws + 1310720);         // 4*32*32*128 f32
    float* ctx2   = (float*)(ws + 3407872);         // 4*32*32*128 f32
    float* xr     = (float*)(ws + 5505024);         // 4*256*256*128 f32 (attn out, shifted)
    float* out    = (float*)d_out;

    k_cvt_bf16<<<192, 256, 0, stream>>>(qkv_w,  qkvw_h, 49152);
    k_cvt_bf16<<<64,  256, 0, stream>>>(proj_w, projw_h, 16384);
    k_cvt_bf16<<<128, 256, 0, stream>>>(fc1_w,  fc1w_h, 32768);
    k_cvt_bf16<<<128, 256, 0, stream>>>(fc2_w,  fc2w_h, 32768);

    k_uni<<<1024, 256, 0, stream>>>(x, uni_w, uni_b, uni);
    k_ngram<<<8192, 64, 0, stream>>>(uni, ng_qkv_w, ng_qkv_b, ng_proj_w, ng_proj_b, ng_bias, ctx);
    k_merge<<<2048, 256, 0, stream>>>(ctx, merge_w, merge_b, ctx2);
    k_wattn<<<4096, 256, 0, stream>>>(x, ctx2, xr, qkvw_h, qkv_b, projw_h, proj_b, btab);
    k_ffn<<<4096, 256, 0, stream>>>(x, xr, out, fc1w_h, fc1_b, fc2w_h, fc2_b,
                                    n1_g, n1_b, n2_g, n2_b);
}
